// EfficientAttention_37383395345086
// MI455X (gfx1250) — compile-verified, matrix operand order checked by reading, not run on hardware
//
#include <hip/hip_runtime.h>
#include <hip/hip_bf16.h>

// ---------------- problem constants ----------------
#define NB      8
#define CCH     512
#define HWP     4096     // H*W
#define NHEADS  8
#define DH      64       // channels per head

typedef __attribute__((ext_vector_type(16))) __bf16 v16bf;
typedef __attribute__((ext_vector_type(8)))  float  v8f;

// ---------------- small helpers ----------------
__device__ __forceinline__ unsigned short f2bf(float f) {
    union { float f; unsigned int u; } x; x.f = f;
    unsigned int u = x.u;
    unsigned int r = u + 0x7FFFu + ((u >> 16) & 1u);   // round-to-nearest-even
    return (unsigned short)(r >> 16);
}
__device__ __forceinline__ float bf2f(unsigned short s) {
    union { unsigned int u; float f; } x; x.u = ((unsigned int)s) << 16;
    return x.f;
}
__device__ __forceinline__ __bf16 bfbits(unsigned short s) {
    union { unsigned short u; __bf16 b; } x; x.u = s; return x.b;
}

// ---------------- f32 -> bf16 conversion ----------------
__global__ void cvt_f32_bf16(const float* __restrict__ in,
                             unsigned short* __restrict__ out, long long n) {
    long long i = (long long)blockIdx.x * blockDim.x + threadIdx.x;
    long long stride = (long long)gridDim.x * blockDim.x;
    for (; i < n; i += stride) out[i] = f2bf(in[i]);
}

// ---------------- WMMA fragment load from LDS ----------------
// LDS tiles are stored with row stride 64 halves (128B). 16-bit A/B 16x32
// fragment layout (ISA 7.12.2): lane m = lane&15, hi = lane>>4;
// VGPR r: K = (r>>2)*16 + hi*8 + (r&3)*2 (+1)
// => per fragment exactly two contiguous 16B chunks per lane (2x ds_load_b128).
__device__ __forceinline__ v16bf load_frag(const unsigned short* lds, int rowBase,
                                           int kk, int lane) {
    const int row = rowBase + (lane & 15);
    const int hi8 = (lane >> 4) << 3;              // 0 or 8 halves
    const unsigned short* p = lds + row * 64 + kk + hi8;
    uint4 q0 = *reinterpret_cast<const uint4*>(p);        // K = kk+hi8 .. +7
    uint4 q1 = *reinterpret_cast<const uint4*>(p + 16);   // K = kk+16+hi8 .. +7
    v16bf f;
    unsigned int w;
    w = q0.x; f[0]  = bfbits((unsigned short)w); f[1]  = bfbits((unsigned short)(w >> 16));
    w = q0.y; f[2]  = bfbits((unsigned short)w); f[3]  = bfbits((unsigned short)(w >> 16));
    w = q0.z; f[4]  = bfbits((unsigned short)w); f[5]  = bfbits((unsigned short)(w >> 16));
    w = q0.w; f[6]  = bfbits((unsigned short)w); f[7]  = bfbits((unsigned short)(w >> 16));
    w = q1.x; f[8]  = bfbits((unsigned short)w); f[9]  = bfbits((unsigned short)(w >> 16));
    w = q1.y; f[10] = bfbits((unsigned short)w); f[11] = bfbits((unsigned short)(w >> 16));
    w = q1.z; f[12] = bfbits((unsigned short)w); f[13] = bfbits((unsigned short)(w >> 16));
    w = q1.w; f[14] = bfbits((unsigned short)w); f[15] = bfbits((unsigned short)(w >> 16));
    return f;
}

// ---------------- batched bf16 WMMA GEMM ----------------
// C[bz] = A[bz](MxK) @ B[bz](KxN) [+ bias[m]]     (B NxK row-major if BT)
// CT: store transposed C[n*ldc+m];  CBF16: bf16 output;  RES: out = res + acc*rw
// REQUIREMENTS (all launches satisfy them): K % 64 == 0; all base pointers and
// row strides 16B-aligned. M/N handled with uniform per-row guards.
template<bool BT, bool CT, bool CBF16, bool RES>
__global__ __launch_bounds__(256)
void gemm_bf16_wmma(const unsigned short* __restrict__ A, long long strideA, int lda,
                    const unsigned short* __restrict__ B, long long strideB, int ldb,
                    void* __restrict__ Cv, long long strideC, int ldc,
                    const float* __restrict__ bias,
                    const float* __restrict__ res, long long strideR,
                    const float* __restrict__ rw,
                    int M, int N, int K)
{
    constexpr int BM = 64, BN = 128, BK = 64;
    __shared__ __align__(16) unsigned short ldsA[BM * BK];   // [m][k]
    __shared__ __align__(16) unsigned short ldsB[BN * BK];   // [n][k]  (N-major)

    const int tid  = threadIdx.x;
    const int lane = tid & 31;
    const int wave = tid >> 5;
    const int wm   = (wave >> 2) * 32;   // 0 / 32
    const int wn   = (wave & 3) * 32;    // 0..96
    const int m0   = blockIdx.y * BM;
    const int n0   = blockIdx.x * BN;
    const int bz   = blockIdx.z;

    const unsigned short* Ab = A + (long long)bz * strideA;
    const unsigned short* Bb = B + (long long)bz * strideB;

    // per-thread fill coordinates (all uniform-guarded, fully vectorized)
    const int arow = tid >> 2;               // 0..63
    const int acol = (tid & 3) << 4;         // 0/16/32/48   (16 halves)
    const int btn  = tid >> 1;               // BT: 0..127
    const int btc  = (tid & 1) << 5;         // BT: 0/32     (32 halves)
    const int bkk  = tid >> 2;               // !BT: 0..63
    const int bnn  = (tid & 3) << 5;         // !BT: 0/32/64/96 (32 halves)

    v8f acc[2][2] = {};

    for (int k0 = 0; k0 < K; k0 += BK) {
        // ---- fill A tile: 64x64, 16 contiguous halves / thread ----
        {
            int gm = m0 + arow;
            uint4 v0 = {0, 0, 0, 0}, v1 = {0, 0, 0, 0};
            if (gm < M) {
                const uint4* gp =
                    reinterpret_cast<const uint4*>(Ab + (long long)gm * lda + k0 + acol);
                if (k0 + BK < K) __builtin_prefetch(gp + (BK >> 3), 0, 0);
                v0 = gp[0]; v1 = gp[1];
            }
            uint4* sp = reinterpret_cast<uint4*>(ldsA + arow * BK + acol);
            sp[0] = v0; sp[1] = v1;
        }
        // ---- fill B tile: 128x64 N-major ----
        if (BT) {            // global B is N x K row-major: straight vector copy
            int gn = n0 + btn;
            uint4 v0 = {0,0,0,0}, v1 = {0,0,0,0}, v2 = {0,0,0,0}, v3 = {0,0,0,0};
            if (gn < N) {
                const uint4* gp =
                    reinterpret_cast<const uint4*>(Bb + (long long)gn * ldb + k0 + btc);
                if (k0 + BK < K) __builtin_prefetch(gp + (BK >> 3), 0, 0);
                v0 = gp[0]; v1 = gp[1]; v2 = gp[2]; v3 = gp[3];
            }
            uint4* sp = reinterpret_cast<uint4*>(ldsB + btn * BK + btc);
            sp[0] = v0; sp[1] = v1; sp[2] = v2; sp[3] = v3;
        } else {             // global B is K x N row-major: vector load, b16 scatter
            int gk = k0 + bkk;
            uint4 v0 = {0,0,0,0}, v1 = {0,0,0,0}, v2 = {0,0,0,0}, v3 = {0,0,0,0};
            if (gk < K) {    // always true for K%64==0, kept for safety (uniform)
                const uint4* gp =
                    reinterpret_cast<const uint4*>(Bb + (long long)gk * ldb + n0 + bnn);
                if (k0 + BK < K)
                    __builtin_prefetch(Bb + (long long)(gk + BK) * ldb + n0 + bnn, 0, 0);
                v0 = gp[0]; v1 = gp[1]; v2 = gp[2]; v3 = gp[3];
            }
            unsigned int u[8] = { v0.x, v0.y, v0.z, v0.w, v1.x, v1.y, v1.z, v1.w };
            unsigned int u2[8] = { v2.x, v2.y, v2.z, v2.w, v3.x, v3.y, v3.z, v3.w };
            #pragma unroll
            for (int j = 0; j < 16; ++j)
                ldsB[(bnn + j) * BK + bkk] =
                    (unsigned short)(u[j >> 1] >> ((j & 1) * 16));
            #pragma unroll
            for (int j = 0; j < 16; ++j)
                ldsB[(bnn + 16 + j) * BK + bkk] =
                    (unsigned short)(u2[j >> 1] >> ((j & 1) * 16));
        }
        __syncthreads();

        #pragma unroll
        for (int kk = 0; kk < BK; kk += 32) {
            v16bf afr[2], bfr[2];
            afr[0] = load_frag(ldsA, wm,      kk, lane);
            afr[1] = load_frag(ldsA, wm + 16, kk, lane);
            bfr[0] = load_frag(ldsB, wn,      kk, lane);
            bfr[1] = load_frag(ldsB, wn + 16, kk, lane);
            #pragma unroll
            for (int ti = 0; ti < 2; ++ti)
                #pragma unroll
                for (int tj = 0; tj < 2; ++tj)
                    acc[ti][tj] = __builtin_amdgcn_wmma_f32_16x16x32_bf16(
                        false, afr[ti], false, bfr[tj], (short)0, acc[ti][tj],
                        false, false);
        }
        __syncthreads();
    }

    // ---- epilogue ----
    const int chi = lane >> 4;          // C layout: m = r + 8*hi, n = lane&15
    const int cn  = lane & 15;
    float rwv = 0.0f;
    if (RES) rwv = rw[0];
    #pragma unroll
    for (int ti = 0; ti < 2; ++ti) {
        #pragma unroll
        for (int tj = 0; tj < 2; ++tj) {
            #pragma unroll
            for (int r = 0; r < 8; ++r) {
                int gm = m0 + wm + ti * 16 + r + 8 * chi;
                int gn = n0 + wn + tj * 16 + cn;
                if (gm < M && gn < N) {
                    float v = acc[ti][tj][r];
                    if (bias) v += bias[gm];
                    long long cidx = (long long)bz * strideC +
                        (CT ? ((long long)gn * ldc + gm) : ((long long)gm * ldc + gn));
                    if (RES) {
                        ((float*)Cv)[cidx] =
                            res[(long long)bz * strideR + (long long)gm * ldc + gn] + v * rwv;
                    } else if (CBF16) {
                        ((unsigned short*)Cv)[cidx] = f2bf(v);
                    } else {
                        ((float*)Cv)[cidx] = v;
                    }
                }
            }
        }
    }
}

// ---------------- softmax over spatial positions (rows), in place, bf16 ----------------
__global__ __launch_bounds__(256)
void softmax_rows_inplace(unsigned short* __restrict__ buf, int rowlen) {
    __shared__ float red[256];
    unsigned short* row = buf + (long long)blockIdx.x * rowlen;
    int tid = threadIdx.x;

    float mx = -1e30f;
    for (int i = tid; i < rowlen; i += 256) mx = fmaxf(mx, bf2f(row[i]));
    red[tid] = mx; __syncthreads();
    for (int s = 128; s > 0; s >>= 1) {
        if (tid < s) red[tid] = fmaxf(red[tid], red[tid + s]);
        __syncthreads();
    }
    mx = red[0]; __syncthreads();

    float sum = 0.0f;
    for (int i = tid; i < rowlen; i += 256) sum += expf(bf2f(row[i]) - mx);
    red[tid] = sum; __syncthreads();
    for (int s = 128; s > 0; s >>= 1) {
        if (tid < s) red[tid] += red[tid + s];
        __syncthreads();
    }
    float inv = 1.0f / red[0];

    for (int i = tid; i < rowlen; i += 256)
        row[i] = f2bf(expf(bf2f(row[i]) - mx) * inv);
}

// ---------------- softmax over head channels (d in 0..63, stride HWP), in place ----------------
__global__ void softmax_heads_inplace(unsigned short* __restrict__ buf) {
    long long idx = (long long)blockIdx.x * blockDim.x + threadIdx.x;
    long long total = (long long)NB * NHEADS * HWP;
    if (idx >= total) return;
    int p = (int)(idx % HWP);
    long long nh = idx / HWP;
    unsigned short* base = buf + nh * (long long)(DH * HWP) + p;

    float mx = -1e30f;
    for (int d = 0; d < DH; ++d) mx = fmaxf(mx, bf2f(base[(long long)d * HWP]));
    float sum = 0.0f;
    for (int d = 0; d < DH; ++d) sum += expf(bf2f(base[(long long)d * HWP]) - mx);
    float inv = 1.0f / sum;
    for (int d = 0; d < DH; ++d)
        base[(long long)d * HWP] = f2bf(expf(bf2f(base[(long long)d * HWP]) - mx) * inv);
}

// ---------------- launcher ----------------
extern "C" void kernel_launch(void* const* d_in, const int* in_sizes, int n_in,
                              void* d_out, int out_size, void* d_ws, size_t ws_size,
                              hipStream_t stream)
{
    const float* x_l = (const float*)d_in[0];
    const float* x_g = (const float*)d_in[1];
    // order: k1,q1,v1 (on x_l), k2,q2,v2 (on x_g)
    const float* w[6]    = { (const float*)d_in[2],  (const float*)d_in[4],  (const float*)d_in[6],
                             (const float*)d_in[8],  (const float*)d_in[10], (const float*)d_in[12] };
    const float* bias[6] = { (const float*)d_in[3],  (const float*)d_in[5],  (const float*)d_in[7],
                             (const float*)d_in[9],  (const float*)d_in[11], (const float*)d_in[13] };
    const float* rw = (const float*)d_in[14];

    const long long XSZ = (long long)NB * CCH * HWP;   // 16,777,216 elements
    const long long WSZ = (long long)CCH * CCH;
    const long long CTX = (long long)NB * NHEADS * DH * DH;

    unsigned short* ws   = (unsigned short*)d_ws;
    unsigned short* xb_l = ws;  ws += XSZ;
    unsigned short* xb_g = ws;  ws += XSZ;
    unsigned short* wb   = ws;  ws += 6 * WSZ;
    unsigned short* proj[6];                 // K1,Q1,V1,K2,Q2,V2 (bf16, (n,c,p))
    for (int i = 0; i < 6; ++i) { proj[i] = ws; ws += XSZ; }
    unsigned short* ctxT_a = ws; ws += CTX;  // from (K1,Q1,V2) -> agg2 -> out_g
    unsigned short* ctxT_b = ws; ws += CTX;  // from (K2,Q2,V1) -> agg1 -> out_l

    // 1) f32 -> bf16
    cvt_f32_bf16<<<4096, 256, 0, stream>>>(x_l, xb_l, XSZ);
    cvt_f32_bf16<<<4096, 256, 0, stream>>>(x_g, xb_g, XSZ);
    for (int i = 0; i < 6; ++i)
        cvt_f32_bf16<<<512, 256, 0, stream>>>(w[i], wb + i * WSZ, WSZ);

    // 2) projections: proj = W(512x512) @ X(512x4096) + b, per batch, bf16 out
    {
        dim3 grid(HWP / 128, CCH / 64, NB);
        const unsigned short* src[6] = { xb_l, xb_l, xb_l, xb_g, xb_g, xb_g };
        for (int i = 0; i < 6; ++i) {
            gemm_bf16_wmma<false, false, true, false><<<grid, 256, 0, stream>>>(
                wb + i * WSZ, 0, CCH,
                src[i], XSZ, HWP,
                proj[i], XSZ, HWP,
                bias[i], nullptr, 0, nullptr,
                CCH, HWP, CCH);
        }
    }

    // 3) softmaxes (in place on bf16 projections)
    softmax_rows_inplace<<<NB * CCH, 256, 0, stream>>>(proj[0], HWP);   // K1 over p
    softmax_rows_inplace<<<NB * CCH, 256, 0, stream>>>(proj[3], HWP);   // K2 over p
    {
        long long tot = (long long)NB * NHEADS * HWP;
        int blocks = (int)((tot + 255) / 256);
        softmax_heads_inplace<<<blocks, 256, 0, stream>>>(proj[1]);     // Q1 over d
        softmax_heads_inplace<<<blocks, 256, 0, stream>>>(proj[4]);     // Q2 over d
    }

    // 4) ctx = Ksm(64x4096) @ V^T(4096x64) per (n,h); store transposed bf16 ctxT[v][k]
    {
        dim3 grid(1, 1, NB * NHEADS);
        const long long sNH = (long long)DH * HWP;     // per-(n,h) stride in proj
        // branch a: K1,Q1 attend over V2 -> agg2
        gemm_bf16_wmma<true, true, true, false><<<grid, 256, 0, stream>>>(
            proj[0], sNH, HWP,
            proj[5], sNH, HWP,
            ctxT_a, (long long)DH * DH, DH,
            nullptr, nullptr, 0, nullptr,
            DH, DH, HWP);
        // branch b: K2,Q2 attend over V1 -> agg1
        gemm_bf16_wmma<true, true, true, false><<<grid, 256, 0, stream>>>(
            proj[3], sNH, HWP,
            proj[2], sNH, HWP,
            ctxT_b, (long long)DH * DH, DH,
            nullptr, nullptr, 0, nullptr,
            DH, DH, HWP);
    }

    // 5) out = x + (ctxT(64x64) @ Qsm(64x4096)) * rw   (fused residual, f32 to d_out)
    {
        dim3 grid(HWP / 128, 1, NB * NHEADS);
        const long long sNH = (long long)DH * HWP;
        float* out_l = (float*)d_out;
        float* out_g = (float*)d_out + XSZ;
        // agg1 -> out_l (residual x_l)
        gemm_bf16_wmma<false, false, false, true><<<grid, 256, 0, stream>>>(
            ctxT_b, (long long)DH * DH, DH,
            proj[4], sNH, HWP,
            out_l, sNH, HWP,
            nullptr, x_l, sNH, rw,
            DH, HWP, DH);
        // agg2 -> out_g (residual x_g)
        gemm_bf16_wmma<false, false, false, true><<<grid, 256, 0, stream>>>(
            ctxT_a, (long long)DH * DH, DH,
            proj[1], sNH, HWP,
            out_g, sNH, HWP,
            nullptr, x_g, sNH, rw,
            DH, HWP, DH);
    }
}
